// CRF_26061861552903
// MI455X (gfx1250) — matrix operand order repaired and checked by reading, not compiled, not run on hardware
//
#include <hip/hip_runtime.h>
#include <hip/hip_bf16.h>
#include <hip/hip_fp16.h>

// CRF loss: B=256, S=1024, T=64.
#define B_ 256
#define S_ 1024
#define T_ 64

typedef __attribute__((ext_vector_type(16))) _Float16     v16h;
typedef __attribute__((ext_vector_type(8)))  float        v8f;
typedef __attribute__((ext_vector_type(4)))  unsigned int v4u;
typedef __attribute__((ext_vector_type(8)))  int          v8i;
typedef __attribute__((ext_vector_type(4)))  int          v4i;

union Frag16 { _Float16 h[16]; v16h v; };
union Frag8f { float    f[8];  v8f  v; };

#define LOG2E_ 1.4426950408889634f
#define LN2_   0.6931471805599453f

__device__ __forceinline__ float fast_exp(float x) {
    return __builtin_amdgcn_exp2f(x * LOG2E_);          // v_exp_f32
}
__device__ __forceinline__ float fast_log(float x) {
    return __builtin_amdgcn_logf(x) * LN2_;             // v_log_f32
}

// TDM: async 2D tile load (16 rows x 64 f32, row stride S*T) into LDS.
__device__ __forceinline__ void tdm_load_tile(const float* gptr, unsigned lds_off) {
    unsigned long long ga = (unsigned long long)(uintptr_t)gptr;
    v4u g0;
    g0[0] = 1u;                                          // count=1 (valid D#)
    g0[1] = lds_off;                                     // lds_addr (bytes)
    g0[2] = (unsigned)(ga & 0xFFFFFFFFu);                // global_addr[31:0]
    g0[3] = (unsigned)((ga >> 32) & 0x01FFFFFFu)         // global_addr[56:32]
          | (2u << 30);                                  // type=2 ("image")
    v8i g1;
    g1[0] = (int)(2u << 16);        // workgroup_mask=0, data_size=2 (4 bytes)
    g1[1] = (int)(64u << 16);       // tensor_dim0 = 64      (bits 79:48 low part)
    g1[2] = (int)(16u << 16);       // tensor_dim1 = 16      (bits 111:80 low part)
    g1[3] = (int)(64u << 16);       // tile_dim0   = 64      (bits 127:112)
    g1[4] = (int)16;                // tile_dim1   = 16, tile_dim2 = 0
    g1[5] = (int)(S_ * T_);         // tensor_dim0_stride = 65536 (bits 207:160)
    g1[6] = 0;
    g1[7] = 0;                      // tensor_dim1_stride unused (2D tile)
    v4i z4 = {0, 0, 0, 0};
#if defined(__clang_major__) && __clang_major__ >= 23
    v8i z8 = {0, 0, 0, 0, 0, 0, 0, 0};
    __builtin_amdgcn_tensor_load_to_lds(g0, g1, z4, z4, z8, 0);
#else
    __builtin_amdgcn_tensor_load_to_lds(g0, g1, z4, z4, 0);
#endif
}

// ---------------------------------------------------------------------------
// Forward algorithm, one wave32 per 16-batch group.
//   alpha'[j] = m + log( sum_i exp(alpha[i]-m) * expT[i,j] ) + emit[t,j]
// expT (f16) in registers as 8 WMMA B-fragments; alpha as 4 C/D f32 fragments.
// Per step: 8 x v_wmma_f32_16x16x32_f16; emissions via TDM double buffer.
// ---------------------------------------------------------------------------
__global__ __launch_bounds__(32) void crf_fwd_wmma(
    const float* __restrict__ em,      // [B,S,T]
    const float* __restrict__ trans,   // [T,T]
    const float* __restrict__ startT,  // [T]
    const float* __restrict__ endT,    // [T]
    float* __restrict__ partition)     // [B]
{
    __shared__ _Float16 ldsA[16 * 64];       // A-matrix staging [M][K]
    __shared__ float    emBuf[2][16 * 64];   // TDM double buffer [M][j]
    const int lane = threadIdx.x;            // 0..31 (wave32)
    const int llo  = lane & 15;
    const int hi   = lane >> 4;              // lane half
    const int b0   = blockIdx.x * 16;        // batch-group base

    // --- B fragments: exp(transitions), contiguous-K B layout (N = lane%16, K = 16*hi+e)
    Frag16 bf[4][2];
#pragma unroll
    for (int n = 0; n < 4; ++n)
#pragma unroll
        for (int kc = 0; kc < 2; ++kc)
#pragma unroll
            for (int e = 0; e < 16; ++e) {
                const int K = kc * 32 + (hi ? 16 : 0) + e;   // source tag i
                const int N = n * 16 + llo;                  // dest tag j
                bf[n][kc].h[e] = (_Float16)fast_exp(trans[K * T_ + N]);
            }

    // --- kick TDM for t=1 while we set up alpha0
    tdm_load_tile(em + (size_t)b0 * S_ * T_ + (size_t)1 * T_,
                  (unsigned)(uintptr_t)&emBuf[1][0]);

    // --- alpha0 = startT + em[t=0], directly in C/D layout (M = r+8*hi, N=llo)
    Frag8f c[4];
#pragma unroll
    for (int n = 0; n < 4; ++n)
#pragma unroll
        for (int r = 0; r < 8; ++r) {
            const int M = r + 8 * hi;
            const int j = n * 16 + llo;
            c[n].f[r] = startT[j] + em[(size_t)(b0 + M) * S_ * T_ + j];
        }

    for (int t = 1; t < S_; ++t) {
        // issue DMA for step t+1 into the other buffer (overlaps this step)
        if (t + 1 < S_)
            tdm_load_tile(em + (size_t)b0 * S_ * T_ + (size_t)(t + 1) * T_,
                          (unsigned)(uintptr_t)&emBuf[(t + 1) & 1][0]);

        // --- row maxes over all 64 columns (4 tiles + 16-lane half reduction)
        float mx[8];
#pragma unroll
        for (int r = 0; r < 8; ++r) {
            float m = fmaxf(fmaxf(c[0].f[r], c[1].f[r]), fmaxf(c[2].f[r], c[3].f[r]));
#pragma unroll
            for (int off = 8; off >= 1; off >>= 1)
                m = fmaxf(m, __shfl_xor(m, off, 32));
            mx[r] = m;
        }

        // --- exp(alpha - m) -> LDS as row-major 16x64 f16 (the A matrix)
#pragma unroll
        for (int n = 0; n < 4; ++n)
#pragma unroll
            for (int r = 0; r < 8; ++r) {
                const int M = r + 8 * hi;
                const int j = n * 16 + llo;
                ldsA[M * 64 + j] = (_Float16)fast_exp(c[n].f[r] - mx[r]);
            }
        __syncthreads();

        // --- A fragments, contiguous K = 16*hi+e as in B (a0: K<32, a1: K>=32)
        Frag16 a0, a1;
#pragma unroll
        for (int v = 0; v < 8; ++v)
#pragma unroll
            for (int s2 = 0; s2 < 2; ++s2) {
                const int K = (hi ? 16 : 0) + v * 2 + s2;
                a0.h[v * 2 + s2] = ldsA[llo * 64 + K];
                a1.h[v * 2 + s2] = ldsA[llo * 64 + K + 32];
            }
        __syncthreads();

        // --- 8 WMMAs: c_tile = A(16x64) x expT(64x16-tile)
        v8f acc[4];
#pragma unroll
        for (int n = 0; n < 4; ++n) {
            v8f a = {};
            a = __builtin_amdgcn_wmma_f32_16x16x32_f16(
                    false, a0.v, false, bf[n][0].v, (short)0, a, false, false);
            a = __builtin_amdgcn_wmma_f32_16x16x32_f16(
                    false, a1.v, false, bf[n][1].v, (short)0, a, false, false);
            acc[n] = a;
        }

        // --- wait for this step's emissions DMA, then alpha' = m + log + emit
        if (t + 1 < S_) __builtin_amdgcn_s_wait_tensorcnt(1);
        else            __builtin_amdgcn_s_wait_tensorcnt(0);
        const float* eb = &emBuf[t & 1][0];
#pragma unroll
        for (int n = 0; n < 4; ++n) {
            Frag8f af; af.v = acc[n];
#pragma unroll
            for (int r = 0; r < 8; ++r) {
                const int M = r + 8 * hi;
                const int j = n * 16 + llo;
                c[n].f[r] = mx[r] + fast_log(af.f[r]) + eb[M * 64 + j];
            }
        }
    }

    // --- partition[b] = logsumexp_j(alpha[b,j] + endT[j])
#pragma unroll
    for (int n = 0; n < 4; ++n)
#pragma unroll
        for (int r = 0; r < 8; ++r)
            c[n].f[r] += endT[n * 16 + llo];

    float mx[8], sm[8];
#pragma unroll
    for (int r = 0; r < 8; ++r) {
        float m = fmaxf(fmaxf(c[0].f[r], c[1].f[r]), fmaxf(c[2].f[r], c[3].f[r]));
#pragma unroll
        for (int off = 8; off >= 1; off >>= 1)
            m = fmaxf(m, __shfl_xor(m, off, 32));
        mx[r] = m;
        float s = 0.f;
#pragma unroll
        for (int n = 0; n < 4; ++n)
            s += fast_exp(c[n].f[r] - m);
#pragma unroll
        for (int off = 8; off >= 1; off >>= 1)
            s += __shfl_xor(s, off, 32);
        sm[r] = s;
    }
    if (llo == 0) {
#pragma unroll
        for (int r = 0; r < 8; ++r)
            partition[b0 + r + 8 * hi] = mx[r] + fast_log(sm[r]);
    }
}

// ---------------------------------------------------------------------------
// Gold-path score: one block per batch, 64 threads strided over t.
// ---------------------------------------------------------------------------
__global__ __launch_bounds__(64) void crf_score(
    const float* __restrict__ em, const int* __restrict__ tags,
    const float* __restrict__ trans, const float* __restrict__ startT,
    const float* __restrict__ endT, float* __restrict__ score)
{
    __shared__ float red[64];
    const int b = blockIdx.x, tid = threadIdx.x;
    const int*   tg = tags + (size_t)b * S_;
    const float* eb = em   + (size_t)b * S_ * T_;

    float acc = 0.f;
    for (int t = tid; t < S_; t += 64) {
        const int cur = tg[t];
        acc += eb[(size_t)t * T_ + cur];
        if (t + 1 < S_) acc += trans[cur * T_ + tg[t + 1]];
    }
    red[tid] = acc;
    __syncthreads();
#pragma unroll
    for (int w = 32; w >= 1; w >>= 1) {
        if (tid < w) red[tid] += red[tid + w];
        __syncthreads();
    }
    if (tid == 0) score[b] = red[0] + startT[tg[0]] + endT[tg[S_ - 1]];
}

// ---------------------------------------------------------------------------
// Final: mean(partition - score), deterministic LDS tree reduction.
// ---------------------------------------------------------------------------
__global__ __launch_bounds__(256) void crf_final(
    const float* __restrict__ partition, const float* __restrict__ score,
    float* __restrict__ out)
{
    __shared__ float red[256];
    const int tid = threadIdx.x;
    red[tid] = partition[tid] - score[tid];
    __syncthreads();
#pragma unroll
    for (int w = 128; w >= 1; w >>= 1) {
        if (tid < w) red[tid] += red[tid + w];
        __syncthreads();
    }
    if (tid == 0) out[0] = red[0] * (1.0f / B_);
}

extern "C" void kernel_launch(void* const* d_in, const int* in_sizes, int n_in,
                              void* d_out, int out_size, void* d_ws, size_t ws_size,
                              hipStream_t stream)
{
    (void)in_sizes; (void)n_in; (void)out_size; (void)ws_size;
    const float* em     = (const float*)d_in[0];   // emissions [B,S,T] f32
    const int*   tags   = (const int*)  d_in[1];   // tags [B,S] int
    const float* trans  = (const float*)d_in[2];   // transitions [T,T] f32
    const float* startT = (const float*)d_in[3];   // [T]
    const float* endT   = (const float*)d_in[4];   // [T]

    float* partition = (float*)d_ws;               // [B]
    float* score     = partition + B_;             // [B]

    crf_fwd_wmma<<<B_ / 16, 32, 0, stream>>>(em, trans, startT, endT, partition);
    crf_score  <<<B_, 64, 0, stream>>>(em, tags, trans, startT, endT, score);
    crf_final  <<<1, 256, 0, stream>>>(partition, score, (float*)d_out);
}
